// RandLANET_7069516169464
// MI455X (gfx1250) — matrix operand order repaired and checked by reading, not compiled, not run on hardware
//
#include <hip/hip_runtime.h>
#include <hip/hip_bf16.h>
#include <cstddef>

#define KNB 16

typedef __attribute__((ext_vector_type(16))) __bf16 v16bf;
typedef __attribute__((ext_vector_type(8)))  float  v8f;

__device__ __forceinline__ float bf2f(unsigned short s) {
  union { unsigned u; float f; } v; v.u = ((unsigned)s) << 16; return v.f;
}
__device__ __forceinline__ unsigned short f2bf(float f) {
  union { float f; unsigned u; } v; v.f = f;
  unsigned r = v.u + 0x7FFFu + ((v.u >> 16) & 1u);   // round to nearest even
  return (unsigned short)(r >> 16);
}
__device__ __forceinline__ float lrelu_f(float x) { return x >= 0.f ? x : 0.2f * x; }

// ---------------------------------------------------------------------------
// Fused dense layer: Y = act( (X @ W) * scale + shift ), bf16 in / bf16 out.
// X: [M, Kp] bf16 row-major (Kp multiple of 16, zero padded)
// Wt: [Np, Kp] bf16 (transposed weights, zero padded)
// A layout (16-bit A 16x32, contiguous k): lanes 0-15 -> K 0-15 (u0..7)
//                                        lanes16-31 -> K 16-31
// B layout (16-bit B striping):          lanes 0-15 K 0-15, lanes 16-31 K 16-31
// C/D: VGPR r -> row r (+8 for lanes 16-31), col = lane & 15
// ---------------------------------------------------------------------------

// one 16x16 tile per wave; handles Kp % 32 == 16 tail (hoisted out of loop)
__global__ __launch_bounds__(256) void wmma_dense_kernel(
    const unsigned short* __restrict__ X,
    const unsigned short* __restrict__ Wt,
    const float* __restrict__ scale,
    const float* __restrict__ shift,
    unsigned short* __restrict__ Y,
    int M, int Kp, int Np, int act)
{
  const int lane = threadIdx.x & 31;
  const int wave = threadIdx.x >> 5;
  const int mt = blockIdx.y * 8 + wave;
  if (mt >= (M >> 4)) return;                 // uniform per wave; EXEC stays full
  const int nt = blockIdx.x;
  const int m0 = mt << 4, n0 = nt << 4;
  const int lrow = lane & 15;
  const int hi = lane >> 4;                   // 0 or 1 (half-wave)

  const unsigned* xa = (const unsigned*)(X  + (size_t)(m0 + lrow) * Kp + hi * 16);
  const unsigned* wb = (const unsigned*)(Wt + (size_t)(n0 + lrow) * Kp + hi * 16);
  const unsigned short* wrow = Wt + (size_t)(n0 + lrow) * Kp;

  v8f acc = {0.f, 0.f, 0.f, 0.f, 0.f, 0.f, 0.f, 0.f};
  const int kfull = Kp & ~31;

  for (int k0 = 0; k0 < kfull; k0 += 32) {
    union { v16bf v; unsigned u[8]; } A, Bm;
    A.u[0] = xa[0];  A.u[1] = xa[1];  A.u[2] = xa[2];  A.u[3] = xa[3];
    A.u[4] = xa[4];  A.u[5] = xa[5];  A.u[6] = xa[6];  A.u[7] = xa[7];
#pragma unroll
    for (int j = 0; j < 8; ++j) Bm.u[j] = wb[j];
    acc = __builtin_amdgcn_wmma_f32_16x16x32_bf16(
        false, A.v, false, Bm.v, (short)0, acc, false, false);
    xa += 16; wb += 16;
  }
  if (Kp & 16) {                              // 16-wide K tail (Kp==16 layers)
    union { v16bf v; unsigned u[8]; } A, Bm;
    if (hi == 0) {
      A.u[0] = xa[0]; A.u[1] = xa[1]; A.u[2] = xa[2]; A.u[3] = xa[3];
      A.u[4] = xa[4]; A.u[5] = xa[5]; A.u[6] = xa[6]; A.u[7] = xa[7];
    } else {
      A.u[0] = 0u;    A.u[1] = 0u;    A.u[2] = 0u;    A.u[3] = 0u;
      A.u[4] = 0u;    A.u[5] = 0u;    A.u[6] = 0u;    A.u[7] = 0u;
    }
    if (hi == 0) {
      const unsigned* wt = (const unsigned*)(wrow + kfull);
#pragma unroll
      for (int j = 0; j < 8; ++j) Bm.u[j] = wt[j];
    } else {
#pragma unroll
      for (int j = 0; j < 8; ++j) Bm.u[j] = 0u;
    }
    acc = __builtin_amdgcn_wmma_f32_16x16x32_bf16(
        false, A.v, false, Bm.v, (short)0, acc, false, false);
  }

  const int col = n0 + lrow;
  const float sc = scale[col], sh = shift[col];
#pragma unroll
  for (int r = 0; r < 8; ++r) {
    float v = acc[r] * sc + sh;
    if (act) v = lrelu_f(v);
    Y[(size_t)(m0 + r + hi * 8) * Np + col] = f2bf(v);
  }
}

// 16x32 tile per wave (A loaded once, reused for 2 N-tiles); needs Kp%32==0
__global__ __launch_bounds__(256) void wmma_dense2_kernel(
    const unsigned short* __restrict__ X,
    const unsigned short* __restrict__ Wt,
    const float* __restrict__ scale,
    const float* __restrict__ shift,
    unsigned short* __restrict__ Y,
    int M, int Kp, int Np, int act)
{
  const int lane = threadIdx.x & 31;
  const int wave = threadIdx.x >> 5;
  const int mt = blockIdx.y * 8 + wave;
  if (mt >= (M >> 4)) return;
  const int n0 = blockIdx.x << 5;
  const int m0 = mt << 4;
  const int lrow = lane & 15;
  const int hi = lane >> 4;

  const unsigned* xa = (const unsigned*)(X  + (size_t)(m0 + lrow) * Kp + hi * 16);
  const unsigned* w0 = (const unsigned*)(Wt + (size_t)(n0 + lrow) * Kp + hi * 16);
  const unsigned* w1 = (const unsigned*)(Wt + (size_t)(n0 + 16 + lrow) * Kp + hi * 16);

  v8f acc0 = {0.f, 0.f, 0.f, 0.f, 0.f, 0.f, 0.f, 0.f};
  v8f acc1 = {0.f, 0.f, 0.f, 0.f, 0.f, 0.f, 0.f, 0.f};

  for (int k0 = 0; k0 < Kp; k0 += 32) {
    union { v16bf v; unsigned u[8]; } A, B0, B1;
    A.u[0] = xa[0];  A.u[1] = xa[1];  A.u[2] = xa[2];  A.u[3] = xa[3];
    A.u[4] = xa[4];  A.u[5] = xa[5];  A.u[6] = xa[6];  A.u[7] = xa[7];
#pragma unroll
    for (int j = 0; j < 8; ++j) B0.u[j] = w0[j];
#pragma unroll
    for (int j = 0; j < 8; ++j) B1.u[j] = w1[j];
    acc0 = __builtin_amdgcn_wmma_f32_16x16x32_bf16(
        false, A.v, false, B0.v, (short)0, acc0, false, false);
    acc1 = __builtin_amdgcn_wmma_f32_16x16x32_bf16(
        false, A.v, false, B1.v, (short)0, acc1, false, false);
    xa += 16; w0 += 16; w1 += 16;
  }

  const int col0 = n0 + lrow, col1 = col0 + 16;
  const float sc0 = scale[col0], sh0 = shift[col0];
  const float sc1 = scale[col1], sh1 = shift[col1];
#pragma unroll
  for (int r = 0; r < 8; ++r) {
    const size_t row = (size_t)(m0 + r + hi * 8) * Np;
    float v0 = acc0[r] * sc0 + sh0;
    float v1 = acc1[r] * sc1 + sh1;
    if (act) { v0 = lrelu_f(v0); v1 = lrelu_f(v1); }
    Y[row + col0] = f2bf(v0);
    Y[row + col1] = f2bf(v1);
  }
}

// --------------------------- elementwise kernels ---------------------------

// features [B,Cin,N] f32 -> [B*N, 16] bf16 (transpose + pad)
__global__ __launch_bounds__(256) void prep_features_kernel(
    const float* __restrict__ feat, unsigned short* __restrict__ out,
    int Bn, int N, int Cin)
{
  size_t i = (size_t)blockIdx.x * blockDim.x + threadIdx.x;
  size_t total = (size_t)Bn * N * 16;
  if (i >= total) return;
  int c = (int)(i & 15);
  size_t m = i >> 4;
  int n = (int)(m % N);
  int b = (int)(m / N);
  float v = (c < Cin) ? feat[((size_t)b * Cin + c) * N + n] : 0.f;
  out[i] = f2bf(v);
}

// relative geometry features: [dis, rel(3), tile(3), neigh(3)] pad-> 16, bf16
__global__ __launch_bounds__(256) void geom_kernel(
    const float* __restrict__ xyz, const int* __restrict__ nidx,
    unsigned short* __restrict__ out, int Bn, int N)
{
  size_t i = (size_t)blockIdx.x * blockDim.x + threadIdx.x;
  size_t total = (size_t)Bn * N * KNB;
  if (i >= total) return;
  int k = (int)(i % KNB);
  size_t m = i / KNB;                 // b*N+n
  int b = (int)(m / N);
  int j = nidx[m * KNB + k];
  const float* pc = xyz + m * 3;
  const float* pn = xyz + ((size_t)b * N + j) * 3;
  float tx = pc[0], ty = pc[1], tz = pc[2];
  float nx = pn[0], ny = pn[1], nz = pn[2];
  float rx = tx - nx, ry = ty - ny, rz = tz - nz;
  float d = sqrtf(rx * rx + ry * ry + rz * rz);
  unsigned short* o = out + (m * KNB + k) * 16;
  o[0] = f2bf(d);
  o[1] = f2bf(rx); o[2] = f2bf(ry); o[3] = f2bf(rz);
  o[4] = f2bf(tx); o[5] = f2bf(ty); o[6] = f2bf(tz);
  o[7] = f2bf(nx); o[8] = f2bf(ny); o[9] = f2bf(nz);
#pragma unroll
  for (int c = 10; c < 16; ++c) o[c] = 0;
}

// out[(m*K+k), c] = c<h ? fpt[gather(idx), c] : (c<2h ? fx[(m*K+k), c-h] : 0)
__global__ __launch_bounds__(256) void gather_concat_kernel(
    const unsigned short* __restrict__ fpt, int wpt,
    const unsigned short* __restrict__ fx, int wx,
    const int* __restrict__ nidx,
    unsigned short* __restrict__ out, int wo, int h, int Bn, int N)
{
  size_t i = (size_t)blockIdx.x * blockDim.x + threadIdx.x;
  size_t total = (size_t)Bn * N * KNB * wo;
  if (i >= total) return;
  int c = (int)(i % wo);
  size_t rk = i / wo;                 // (b*N+n)*K + k
  size_t m = rk / KNB;
  int b = (int)(m / N);
  unsigned short v = 0;
  if (c < h) {
    int j = nidx[rk];
    v = fpt[((size_t)b * N + j) * wpt + c];
  } else if (c < 2 * h) {
    v = fx[rk * wx + (c - h)];
  }
  out[i] = v;
}

// softmax over K per (point, channel), weighted sum of F
__global__ __launch_bounds__(256) void att_reduce_kernel(
    const unsigned short* __restrict__ S, const unsigned short* __restrict__ F,
    unsigned short* __restrict__ out, int w, size_t Mrows)
{
  size_t i = (size_t)blockIdx.x * blockDim.x + threadIdx.x;
  size_t total = Mrows * (size_t)w;
  if (i >= total) return;
  int c = (int)(i % w);
  size_t m = i / w;
  const unsigned short* sp = S + m * KNB * (size_t)w + c;
  const unsigned short* fp = F + m * KNB * (size_t)w + c;
  float mx = -3.4e38f;
#pragma unroll
  for (int k = 0; k < KNB; ++k) mx = fmaxf(mx, bf2f(sp[(size_t)k * w]));
  float den = 0.f, num = 0.f;
#pragma unroll
  for (int k = 0; k < KNB; ++k) {
    float e = __expf(bf2f(sp[(size_t)k * w]) - mx);
    den += e;
    num += e * bf2f(fp[(size_t)k * w]);
  }
  out[i] = f2bf(num / den);
}

__global__ __launch_bounds__(256) void add_lrelu_kernel(
    const unsigned short* __restrict__ a, const unsigned short* __restrict__ b,
    unsigned short* __restrict__ o, size_t total)
{
  size_t i = (size_t)blockIdx.x * blockDim.x + threadIdx.x;
  if (i >= total) return;
  o[i] = f2bf(lrelu_f(bf2f(a[i]) + bf2f(b[i])));
}

// out[b,n,c] = max_k src[b, sidx[b,n,k], c]
__global__ __launch_bounds__(256) void gather_max_kernel(
    const unsigned short* __restrict__ src, int w, const int* __restrict__ sidx,
    unsigned short* __restrict__ out, int Bn, int Nout, int Nin)
{
  size_t i = (size_t)blockIdx.x * blockDim.x + threadIdx.x;
  size_t total = (size_t)Bn * Nout * w;
  if (i >= total) return;
  int c = (int)(i % w);
  size_t m = i / w;
  int b = (int)(m / Nout);
  const int* sp = sidx + m * KNB;
  float mx = -3.4e38f;
#pragma unroll
  for (int k = 0; k < KNB; ++k) {
    int j = sp[k];
    mx = fmaxf(mx, bf2f(src[((size_t)b * Nin + j) * w + c]));
  }
  out[i] = f2bf(mx);
}

// nearest-neighbor interpolation gather
__global__ __launch_bounds__(256) void gather_rows_kernel(
    const unsigned short* __restrict__ src, int w, const int* __restrict__ gidx,
    unsigned short* __restrict__ out, int Bn, int Nout, int Nin)
{
  size_t i = (size_t)blockIdx.x * blockDim.x + threadIdx.x;
  size_t total = (size_t)Bn * Nout * w;
  if (i >= total) return;
  int c = (int)(i % w);
  size_t m = i / w;
  int b = (int)(m / Nout);
  int j = gidx[m];
  out[i] = src[((size_t)b * Nin + j) * w + c];
}

__global__ __launch_bounds__(256) void concat2_kernel(
    const unsigned short* __restrict__ A, int wa,
    const unsigned short* __restrict__ Bs, int wb,
    unsigned short* __restrict__ out, int wo, size_t Mrows)
{
  size_t i = (size_t)blockIdx.x * blockDim.x + threadIdx.x;
  size_t total = Mrows * (size_t)wo;
  if (i >= total) return;
  int c = (int)(i % wo);
  size_t m = i / wo;
  unsigned short v = 0;
  if (c < wa) v = A[m * wa + c];
  else if (c < wa + wb) v = Bs[m * wb + (c - wa)];
  out[i] = v;
}

// W [din,dout] f32 -> Wt [Np,Kp] bf16 transposed + zero padded
__global__ __launch_bounds__(256) void weight_cvt_kernel(
    const float* __restrict__ W, unsigned short* __restrict__ Wt,
    int din, int dout, int Kp, int Np)
{
  size_t i = (size_t)blockIdx.x * blockDim.x + threadIdx.x;
  size_t total = (size_t)Kp * Np;
  if (i >= total) return;
  int k = (int)(i % Kp);
  int col = (int)(i / Kp);
  float v = (k < din && col < dout) ? W[(size_t)k * dout + col] : 0.f;
  Wt[i] = f2bf(v);
}

// fold bias + eval-BN into per-channel scale/shift (zero in pad lanes)
__global__ __launch_bounds__(256) void make_scale_kernel(
    const float* __restrict__ b, const float* __restrict__ g,
    const float* __restrict__ bt, float* __restrict__ sc, float* __restrict__ sh,
    int dout, int Np, float bns)
{
  int c = blockIdx.x * blockDim.x + threadIdx.x;
  if (c >= Np) return;
  if (c >= dout) { sc[c] = 0.f; sh[c] = 0.f; return; }
  if (g) { float a = g[c] * bns; sc[c] = a; sh[c] = b[c] * a + bt[c]; }
  else   { sc[c] = 1.f; sh[c] = b[c]; }
}

// [B*N, w] bf16 -> d_out [B, ncls, N] f32
__global__ __launch_bounds__(256) void emit_out_kernel(
    const unsigned short* __restrict__ Y, float* __restrict__ out,
    int Bn, int N, int ncls, int w)
{
  size_t i = (size_t)blockIdx.x * blockDim.x + threadIdx.x;
  size_t total = (size_t)Bn * ncls * N;
  if (i >= total) return;
  int n = (int)(i % N);
  size_t t2 = i / N;
  int cls = (int)(t2 % ncls);
  int b = (int)(t2 / ncls);
  out[i] = bf2f(Y[((size_t)b * N + n) * w + cls]);
}

// ------------------------------- host side ---------------------------------

static inline int cp16(int x) { return (x + 15) & ~15; }

namespace {
struct Bump {
  char* p; size_t off;
  void* take(size_t n) { void* r = p + off; off = (off + n + 255) & ~(size_t)255; return r; }
};
struct DenseL { unsigned short* Wt; float* sc; float* sh; int Kp, Np; };
}

extern "C" void kernel_launch(void* const* d_in, const int* in_sizes, int n_in,
                              void* d_out, int out_size, void* d_ws, size_t ws_size,
                              hipStream_t stream) {
  (void)in_sizes; (void)n_in; (void)out_size; (void)ws_size;
  static const int NSh[5] = {40960, 10240, 2560, 640, 160};
  static const int DO[4]  = {16, 64, 128, 256};
  static const int DIN[4] = {8, 32, 128, 256};
  const int Bn = 4, NCLS = 13;
  const float BNS = 0.9999950000374997f;     // 1/sqrt(1+1e-5)

  Bump ws{(char*)d_ws, 0};

  // ---- parse param leaves (after 20 data inputs; jax sorted-key order) ----
  int pi = 20;
  auto takeL = [&](int din, int dout, bool bn) -> DenseL {
    const float* W  = (const float*)d_in[pi++];
    const float* bb = (const float*)d_in[pi++];
    const float* bt = nullptr; const float* g = nullptr;
    if (bn) { bt = (const float*)d_in[pi++]; g = (const float*)d_in[pi++]; }
    DenseL L; L.Kp = cp16(din); L.Np = cp16(dout);
    L.Wt = (unsigned short*)ws.take((size_t)L.Kp * L.Np * 2);
    L.sc = (float*)ws.take((size_t)L.Np * 4);
    L.sh = (float*)ws.take((size_t)L.Np * 4);
    size_t tot = (size_t)L.Kp * L.Np;
    weight_cvt_kernel<<<(unsigned)((tot + 255) / 256), 256, 0, stream>>>(W, L.Wt, din, dout, L.Kp, L.Np);
    make_scale_kernel<<<(unsigned)((L.Np + 255) / 256), 256, 0, stream>>>(bb, g, bt, L.sc, L.sh, dout, L.Np, BNS);
    return L;
  };

  DenseL dec[4];
  const int decd[4][2] = {{768, 256}, {384, 128}, {160, 32}, {64, 32}};
  for (int j = 0; j < 4; ++j) dec[j] = takeL(decd[j][0], decd[j][1], true);
  DenseL dec0 = takeL(512, 512, true);
  struct EncL { DenseL a1fc, a1mlp, a2fc, a2mlp, l1, l2, m1, m2, sc; } enc[4];
  for (int i = 0; i < 4; ++i) {
    int d = DO[i], h = d / 2;
    enc[i].a1fc  = takeL(d, d, false);
    enc[i].a1mlp = takeL(d, h, true);
    enc[i].a2fc  = takeL(d, d, false);
    enc[i].a2mlp = takeL(d, d, true);
    enc[i].l1    = takeL(10, h, true);
    enc[i].l2    = takeL(h, h, true);
    enc[i].m1    = takeL(DIN[i], h, true);
    enc[i].m2    = takeL(d, 2 * d, true);
    enc[i].sc    = takeL(DIN[i], 2 * d, true);
  }
  DenseL fc0l = takeL(6, 8, true), fc1l = takeL(32, 64, true),
         fc2l = takeL(64, 32, true), fc3l = takeL(32, 13, false);

  auto runDense = [&](const DenseL& L, const unsigned short* X, unsigned short* Y,
                      size_t M, int act) {
    if ((L.Np & 31) == 0 && (L.Kp & 31) == 0) {
      dim3 grid((unsigned)(L.Np / 32), (unsigned)((M / 16 + 7) / 8));
      wmma_dense2_kernel<<<grid, 256, 0, stream>>>(X, L.Wt, L.sc, L.sh, Y, (int)M, L.Kp, L.Np, act);
    } else {
      dim3 grid((unsigned)(L.Np / 16), (unsigned)((M / 16 + 7) / 8));
      wmma_dense_kernel<<<grid, 256, 0, stream>>>(X, L.Wt, L.sc, L.sh, Y, (int)M, L.Kp, L.Np, act);
    }
  };

  // ---- activation arenas (bf16, channel-padded to 16) ----
  size_t Ms[5]; for (int i = 0; i < 5; ++i) Ms[i] = (size_t)Bn * NSh[i];
  size_t gslot = 0, s_f1 = 0, s_wd = 0, s_w2 = 0;
  for (int i = 0; i < 4; ++i) {
    size_t MK = Ms[i] * KNB;
    size_t a = MK * cp16(DO[i]) * 2;            if (a > gslot) gslot = a;
    size_t b1 = Ms[i] * cp16(DO[i] / 2) * 2;    if (b1 > s_f1) s_f1 = b1;
    size_t b2 = Ms[i] * cp16(DO[i]) * 2;        if (b2 > s_wd) s_wd = b2;
    size_t b3 = Ms[i] * cp16(2 * DO[i]) * 2;    if (b3 > s_w2) s_w2 = b3;
  }
  unsigned short* G0 = (unsigned short*)ws.take(gslot);   // raw -> scores1 -> fxyz2
  unsigned short* G1 = (unsigned short*)ws.take(gslot);   // fxyz -> scores2
  unsigned short* G3 = (unsigned short*)ws.take(gslot);   // fcat1 -> fcat2
  unsigned short* Pf1      = (unsigned short*)ws.take(s_f1);
  unsigned short* PaggPre  = (unsigned short*)ws.take(s_wd);
  unsigned short* Pagg     = (unsigned short*)ws.take(s_f1);
  unsigned short* PlfaPre  = (unsigned short*)ws.take(s_wd);
  unsigned short* Pflfa    = (unsigned short*)ws.take(s_wd);
  unsigned short* Pmain    = (unsigned short*)ws.take(s_w2);
  unsigned short* Pshort   = (unsigned short*)ws.take(s_w2);
  unsigned short* Pfe      = (unsigned short*)ws.take(s_w2);
  unsigned short* fin0 = (unsigned short*)ws.take(Ms[0] * 16 * 2);
  unsigned short* f0   = (unsigned short*)ws.take(Ms[0] * 16 * 2);
  unsigned short* fe0  = (unsigned short*)ws.take(Ms[0] * 32 * 2);
  unsigned short* fsb[4];
  for (int i = 0; i < 4; ++i) fsb[i] = (unsigned short*)ws.take(Ms[i + 1] * cp16(2 * DO[i]) * 2);
  unsigned short* fd = (unsigned short*)ws.take(Ms[4] * 512 * 2);
  const int curws[4]  = {512, 256, 128, 32};
  const int skipws[4] = {256, 128, 32, 32};
  size_t fiSlot = 0, catSlot = 0;
  for (int j = 0; j < 4; ++j) {
    int lvl = 3 - j;
    size_t a = Ms[lvl] * (size_t)curws[j] * 2;                 if (a > fiSlot)  fiSlot  = a;
    size_t c = Ms[lvl] * (size_t)(curws[j] + skipws[j]) * 2;   if (c > catSlot) catSlot = c;
  }
  unsigned short* Fi  = (unsigned short*)ws.take(fiSlot);
  unsigned short* Cat = (unsigned short*)ws.take(catSlot);
  unsigned short* Da  = (unsigned short*)ws.take(Ms[0] * 64 * 2);
  unsigned short* Db  = (unsigned short*)ws.take(Ms[0] * 64 * 2);

  auto blocks = [](size_t t) { return (unsigned)((t + 255) / 256); };

  // ---- fc0 ----
  {
    size_t t = Ms[0] * 16;
    prep_features_kernel<<<blocks(t), 256, 0, stream>>>((const float*)d_in[16], fin0, Bn, NSh[0], 6);
    runDense(fc0l, fin0, f0, Ms[0], 1);
  }

  // ---- encoder ----
  unsigned short* fin = f0;
  for (int i = 0; i < 4; ++i) {
    const float* xyz = (const float*)d_in[i * 4 + 0];
    const int* nidx  = (const int*)d_in[i * 4 + 1];
    const int* sidx  = (const int*)d_in[i * 4 + 2];
    int d = DO[i], h = d / 2;
    int wh = cp16(h), wd = cp16(d), w2 = cp16(2 * d);
    size_t Mi = Ms[i], MK = Mi * KNB;
    size_t t;

    t = MK;      geom_kernel<<<blocks(t), 256, 0, stream>>>(xyz, nidx, G0, Bn, NSh[i]);
    runDense(enc[i].l1, G0, G1, MK, 1);                              // f_xyz
    runDense(enc[i].m1, fin, Pf1, Mi, 1);                            // f = mlp1(feat)
    t = MK * wd; gather_concat_kernel<<<blocks(t), 256, 0, stream>>>(Pf1, wh, G1, wh, nidx, G3, wd, h, Bn, NSh[i]);
    runDense(enc[i].a1fc, G3, G0, MK, 0);                            // att scores 1
    t = Mi * wd; att_reduce_kernel<<<blocks(t), 256, 0, stream>>>(G0, G3, PaggPre, wd, Mi);
    runDense(enc[i].a1mlp, PaggPre, Pagg, Mi, 1);                    // agg
    runDense(enc[i].l2, G1, G0, MK, 1);                              // f_xyz2
    t = MK * wd; gather_concat_kernel<<<blocks(t), 256, 0, stream>>>(Pagg, wh, G0, wh, nidx, G3, wd, h, Bn, NSh[i]);
    runDense(enc[i].a2fc, G3, G1, MK, 0);                            // att scores 2
    t = Mi * wd; att_reduce_kernel<<<blocks(t), 256, 0, stream>>>(G1, G3, PlfaPre, wd, Mi);
    runDense(enc[i].a2mlp, PlfaPre, Pflfa, Mi, 1);                   // lfa out
    runDense(enc[i].m2, Pflfa, Pmain, Mi, 1);                        // mlp2
    runDense(enc[i].sc, fin, Pshort, Mi, 0);                         // shortcut
    unsigned short* fe = (i == 0) ? fe0 : Pfe;
    t = Mi * w2; add_lrelu_kernel<<<blocks(t), 256, 0, stream>>>(Pmain, Pshort, fe, t);
    t = Ms[i + 1] * w2;
    gather_max_kernel<<<blocks(t), 256, 0, stream>>>(fe, w2, sidx, fsb[i], Bn, NSh[i + 1], NSh[i]);
    fin = fsb[i];
  }

  // ---- decoder ----
  runDense(dec0, fsb[3], fd, Ms[4], 1);
  unsigned short* cur = fd; int curw = 512;
  unsigned short* skips[4] = {fsb[2], fsb[1], fsb[0], fe0};
  unsigned short* outs[4]  = {Da, Db, Da, Db};
  for (int j = 0; j < 4; ++j) {
    int lvl = 3 - j;
    const int* gidx = (const int*)d_in[lvl * 4 + 3];
    size_t Mo = Ms[lvl];
    size_t t = Mo * (size_t)curw;
    gather_rows_kernel<<<blocks(t), 256, 0, stream>>>(cur, curw, gidx, Fi, Bn, NSh[lvl], NSh[lvl + 1]);
    int sw = skipws[j], wo = sw + curw;
    t = Mo * (size_t)wo;
    concat2_kernel<<<blocks(t), 256, 0, stream>>>(skips[j], sw, Fi, curw, Cat, wo, Mo);
    runDense(dec[j], Cat, outs[j], Mo, 1);
    cur = outs[j]; curw = dec[j].Np;
  }

  // ---- head ----
  runDense(fc1l, cur, Da, Ms[0], 1);     // cur == Db (32ch)
  runDense(fc2l, Da, Db, Ms[0], 1);
  runDense(fc3l, Db, Da, Ms[0], 0);
  size_t t = (size_t)Bn * NCLS * NSh[0];
  emit_out_kernel<<<blocks(t), 256, 0, stream>>>(Da, (float*)d_out, Bn, NSh[0], NCLS, 13 == NCLS ? 16 : 16);
}
